// PDEANODE_46823733461190
// MI455X (gfx1250) — compile-verified, matrix operand order checked by reading, not run on hardware
//
#include <hip/hip_runtime.h>
#include <hip/hip_bf16.h>

typedef __attribute__((ext_vector_type(16))) _Float16 v16h;
typedef __attribute__((ext_vector_type(8)))  float    v8f;
typedef __attribute__((ext_vector_type(4)))  unsigned v4u;

#define NB      256   // batch
#define NT      64    // saved time points
#define NDATA   256   // data dims
#define NAUG    8
#define ND      264   // D = DATA + AUG
#define NW      256   // hidden width
#define KPAD    288   // K padded to 9*32 for layer-0 input / layer-3 weight rows
#define ROWS    16    // batch rows per workgroup (one WMMA M-tile)
#define THREADS 512   // 16 wave32 waves
#define NSUBS   4
#define NPAIRS  (ROWS * ND / 2)   // 2112 float pairs of state per workgroup
#define PPT     5                 // pairs per thread (ceil(2112/512))

// f16 padded weight layout in d_ws (offsets in halves), all row-major [out][in], stride KPAD
#define OFF_W0  0
#define OFF_W1  (256 * KPAD)
#define OFF_W2  (2 * 256 * KPAD)
#define OFF_W3  (3 * 256 * KPAD)
#define WH_TOTAL (3 * 256 * KPAD + KPAD * KPAD)

// Branch-free tanh: clamp, v_exp_f32 (2^x), v_rcp_f32. tanh(x) = (e^{2x}-1)/(e^{2x}+1).
__device__ __forceinline__ float tanh_fast(float x) {
  x = fminf(9.0f, fmaxf(-9.0f, x));
  const float e = __builtin_amdgcn_exp2f(x * 2.8853900817779268f); // e^(2x)
  return (e - 1.0f) * __builtin_amdgcn_rcpf(e + 1.0f);
}

__device__ __forceinline__ unsigned pack2h(float a, float b) {
  union { unsigned u; _Float16 h[2]; } p;
  p.h[0] = (_Float16)a; p.h[1] = (_Float16)b;
  return p.u;
}

// ---- B-fragment loads via inline asm: opaque to LICM so the compiler can NEVER
// hoist the loop-invariant weight fragments into (spilled) registers. GVS mode:
// uniform SGPR base + per-lane byte offset; the fragment is two b128 chunks at +0/+32.
__device__ __forceinline__ void bload(const _Float16* __restrict__ base,
                                      unsigned voff, v4u& lo, v4u& hi) {
  asm volatile("global_load_b128 %0, %2, %3\n\t"
               "global_load_b128 %1, %2, %3 offset:32"
               : "=&v"(lo), "=&v"(hi)
               : "v"(voff), "s"(base)
               : "memory");
}
// Wait for the fragment's loads (in-order return; CNT=2 tolerates one prefetched
// fragment in flight). Ties the data regs so WMMA can't be scheduled before the wait.
template <int CNT>
__device__ __forceinline__ void bwait(v4u& lo, v4u& hi) {
  if constexpr (CNT == 0)
    asm volatile("s_wait_loadcnt 0x0" : "+v"(lo), "+v"(hi));
  else
    asm volatile("s_wait_loadcnt 0x2" : "+v"(lo), "+v"(hi));
}
__device__ __forceinline__ v16h bcombine(const v4u& lo, const v4u& hi) {
  union { v4u q[2]; v16h v; } u;
  u.q[0] = lo; u.q[1] = hi;
  return u.v;
}

// A-fragment (activations) from LDS. Per ISA 7.12.2 (16-bit A 16x32): lane = 16*kh + r;
// per lane two contiguous 16B chunks -> lowers to paired ds_load_b128.
__device__ __forceinline__ v16h load_a_frag(const _Float16* base, int stride,
                                            int k0, int lane) {
  const int r  = lane & 15;
  const int kh = (lane >> 4) << 3;
  const _Float16* p = base + r * stride;
  union { v16h v; unsigned u[8]; } f;
#pragma unroll
  for (int i = 0; i < 8; ++i) {
    const int kk = k0 + ((i < 4) ? (kh + 2 * i) : (16 + kh + 2 * (i - 4)));
    f.u[i] = *(const unsigned*)(p + kk);
  }
  return f.v;
}

__global__ void prep_weights(const float* __restrict__ W0, const float* __restrict__ W1,
                             const float* __restrict__ W2, const float* __restrict__ W3,
                             _Float16* __restrict__ wh) {
  int idx = blockIdx.x * blockDim.x + threadIdx.x;
  if (idx >= WH_TOTAL) return;
  float v = 0.0f;
  if (idx < OFF_W1) {                       // W0: [256][264] -> [256][288]
    int o = idx / KPAD, k = idx % KPAD;
    if (k < ND) v = W0[o * ND + k];
  } else if (idx < OFF_W2) {                // W1: [256][256] -> [256][288]
    int r = idx - OFF_W1; int o = r / KPAD, k = r % KPAD;
    if (k < NW) v = W1[o * NW + k];
  } else if (idx < OFF_W3) {                // W2: [256][256] -> [256][288]
    int r = idx - OFF_W2; int o = r / KPAD, k = r % KPAD;
    if (k < NW) v = W2[o * NW + k];
  } else {                                  // W3: [264][256] -> [288][288]
    int r = idx - OFF_W3; int o = r / KPAD, k = r % KPAD;
    if (o < ND && k < NW) v = W3[o * NW + k];
  }
  wh[idx] = (_Float16)v;
}

__global__ __launch_bounds__(THREADS)
void node_kernel(const float* __restrict__ ts, const float* __restrict__ xs,
                 const float* __restrict__ a_s,
                 const float* __restrict__ b0, const float* __restrict__ b1,
                 const float* __restrict__ b2, const float* __restrict__ b3,
                 const _Float16* __restrict__ wh, float* __restrict__ out) {
  __shared__ float    kf[ROWS * ND];      // f() output, f32
  __shared__ _Float16 a16[ROWS * KPAD];   // f input / L1 output (f16, K-padded)
  __shared__ _Float16 h16[ROWS * NW];     // L0 / L2 output (f16)

  const int tid     = threadIdx.x;
  const int lane    = tid & 31;
  const int wv      = tid >> 5;           // wave id 0..15 == N-tile id
  const int rowBase = blockIdx.x * ROWS;
  const int n       = lane & 15;
  const int mh      = (lane >> 4) * 8;
  // per-lane byte offset of this lane's B-fragment row for tile base t: ((16t+n)*KPAD + kh)*2
  const int khh     = (lane >> 4) << 3;   // half offset 0 or 8

  // Wave-constant biases (hoisted out of the time loop).
  const float bs0 = b0[16 * wv + n];
  const float bs1 = b1[16 * wv + n];
  const float bs2 = b2[16 * wv + n];
  const float bs3 = b3[16 * wv + n];                         // 16*wv+n <= 255 < 264
  const float bs3b = (wv == 0 && 256 + n < ND) ? b3[256 + n] : 0.0f;

  // ---- per-thread element ownership: PPT float-pairs of the [16 x 264] state ----
  int   cc[PPT];                    // column (even) or -1 if pair unused
  int   idxA[PPT];                  // a16 offset (halves)
  int   idxK[PPT];                  // kf  offset (floats)
  long  obase[PPT];                 // out base offset for this pair
  float yv[PPT][2];                 // master state y, registers
  float k1[PPT][2];                 // Bosh3 accumulator, registers

  // zero a16 once (K-pad columns 264..287 must stay 0 forever)
  for (int i = tid; i < ROWS * KPAD; i += THREADS) a16[i] = (_Float16)0.0f;

  const float sc_t = expf(-0.1f * ts[0]);
#pragma unroll
  for (int j = 0; j < PPT; ++j) {
    const int i2 = tid + THREADS * j;
    if (i2 < NPAIRS) {
      const int m = i2 / (ND / 2);
      const int c = (i2 % (ND / 2)) * 2;
      cc[j]    = c;
      idxA[j]  = m * KPAD + c;
      idxK[j]  = m * ND + c;
      obase[j] = (long)(rowBase + m) * NT * NDATA + c;
      if (c < NDATA) {
        const float sc = a_s[rowBase + m] * sc_t;
        yv[j][0] = sc * sinf(3.14159265358979323846f * xs[(rowBase + m) * NDATA + c]);
        yv[j][1] = sc * sinf(3.14159265358979323846f * xs[(rowBase + m) * NDATA + c + 1]);
      } else {
        yv[j][0] = yv[j][1] = 0.0f;               // augmented dims start at zero
      }
    } else {
      cc[j] = -1; idxA[j] = 0; idxK[j] = 0; obase[j] = 0;
      yv[j][0] = yv[j][1] = 0.0f;
    }
    k1[j][0] = k1[j][1] = 0.0f;
  }

  // t = 0 output slice straight from registers
#pragma unroll
  for (int j = 0; j < PPT; ++j)
    if (cc[j] >= 0 && cc[j] < NDATA)
      *(float2*)(out + obase[j]) = make_float2(yv[j][0], yv[j][1]);

  // One GEMM layer: out16x16 = A(LDS) @ Wt^T with software-pipelined asm B loads.
  // KT = number of 32-wide K tiles; tileRow = N-tile row base in the weight matrix.
  auto gemm = [&](const _Float16* aBase, int aStride, const _Float16* wBase,
                  int tileRow, int KT) -> v8f {
    const unsigned voff = (unsigned)(((tileRow + n) * KPAD + khh) * 2);
    v8f c = {0.f, 0.f, 0.f, 0.f, 0.f, 0.f, 0.f, 0.f};
    v4u blo[2], bhi[2];
    bload(wBase, voff, blo[0], bhi[0]);
#pragma unroll
    for (int k = 0; k < 9; ++k) {           // max KT = 9; guarded below
      if (k >= KT) break;
      if (k + 1 < KT)
        bload(wBase, voff + 64u * (unsigned)(k + 1), blo[(k + 1) & 1], bhi[(k + 1) & 1]);
      if (k + 1 < KT) bwait<2>(blo[k & 1], bhi[k & 1]);
      else            bwait<0>(blo[k & 1], bhi[k & 1]);
      v16h a = load_a_frag(aBase, aStride, 32 * k, lane);
      v16h b = bcombine(blo[k & 1], bhi[k & 1]);
      c = __builtin_amdgcn_wmma_f32_16x16x32_f16(false, a, false, b, (short)0, c,
                                                 false, false);
    }
    return c;
  };

  // ---- vector field: kf = MLP(a16); f16 weights stream from L2 every call ----
  auto runf = [&]() {
    { // L0: a16 (K=288) -> tanh -> h16
      v8f c = gemm(a16, KPAD, wh + OFF_W0, 16 * wv, KPAD / 32);
#pragma unroll
      for (int r = 0; r < 8; ++r)
        h16[(mh + r) * NW + 16 * wv + n] = (_Float16)tanh_fast(c[r] + bs0);
    }
    __syncthreads();
    { // L1: h16 (K=256) -> tanh -> a16 (ping-pong; f-input dead now)
      v8f c = gemm(h16, NW, wh + OFF_W1, 16 * wv, NW / 32);
#pragma unroll
      for (int r = 0; r < 8; ++r)
        a16[(mh + r) * KPAD + 16 * wv + n] = (_Float16)tanh_fast(c[r] + bs1);
    }
    __syncthreads();
    { // L2: a16 (K=256) -> tanh -> h16
      v8f c = gemm(a16, KPAD, wh + OFF_W2, 16 * wv, NW / 32);
#pragma unroll
      for (int r = 0; r < 8; ++r)
        h16[(mh + r) * NW + 16 * wv + n] = (_Float16)tanh_fast(c[r] + bs2);
    }
    __syncthreads();
    { // L3: h16 (K=256) -> linear -> kf f32. 17 N-tiles: every wave does tile wv,
      // wave 0 additionally does tile 16 (wave-uniform branch; EXEC stays full).
      v8f c = gemm(h16, NW, wh + OFF_W3, 16 * wv, NW / 32);
#pragma unroll
      for (int r = 0; r < 8; ++r)
        kf[(mh + r) * ND + 16 * wv + n] = c[r] + bs3;
      if (wv == 0) {
        v8f c2 = gemm(h16, NW, wh + OFF_W3, 256, NW / 32);
        const int col = 256 + n;
        if (col < ND) {
#pragma unroll
          for (int r = 0; r < 8; ++r)
            kf[(mh + r) * ND + col] = c2[r] + bs3b;
        }
      }
    }
    __syncthreads();
  };

  for (int stp = 0; stp < NT - 1; ++stp) {
    const float hh = (ts[stp + 1] - ts[stp]) * (1.0f / NSUBS);
    for (int sub = 0; sub < NSUBS; ++sub) {
      // stage y -> a16 (f16 pairs)
#pragma unroll
      for (int j = 0; j < PPT; ++j)
        if (cc[j] >= 0) *(unsigned*)&a16[idxA[j]] = pack2h(yv[j][0], yv[j][1]);
      __syncthreads();
      runf();                               // kf = k1
      // save k1; stage y + 0.5*h*k1
#pragma unroll
      for (int j = 0; j < PPT; ++j)
        if (cc[j] >= 0) {
          const float2 kv = *(const float2*)&kf[idxK[j]];
          k1[j][0] = kv.x; k1[j][1] = kv.y;
          *(unsigned*)&a16[idxA[j]] =
              pack2h(yv[j][0] + 0.5f * hh * kv.x, yv[j][1] + 0.5f * hh * kv.y);
        }
      __syncthreads();
      runf();                               // kf = k2
      // acc = 2/9 k1 + 1/3 k2; stage y + 0.75*h*k2
#pragma unroll
      for (int j = 0; j < PPT; ++j)
        if (cc[j] >= 0) {
          const float2 kv = *(const float2*)&kf[idxK[j]];
          k1[j][0] = (2.0f / 9.0f) * k1[j][0] + (1.0f / 3.0f) * kv.x;
          k1[j][1] = (2.0f / 9.0f) * k1[j][1] + (1.0f / 3.0f) * kv.y;
          *(unsigned*)&a16[idxA[j]] =
              pack2h(yv[j][0] + 0.75f * hh * kv.x, yv[j][1] + 0.75f * hh * kv.y);
        }
      __syncthreads();
      runf();                               // kf = k3
      // y += h*(acc + 4/9 k3)   (registers only; no barrier needed before next stage)
#pragma unroll
      for (int j = 0; j < PPT; ++j)
        if (cc[j] >= 0) {
          const float2 kv = *(const float2*)&kf[idxK[j]];
          yv[j][0] += hh * (k1[j][0] + (4.0f / 9.0f) * kv.x);
          yv[j][1] += hh * (k1[j][1] + (4.0f / 9.0f) * kv.y);
        }
    }
    // save interval-end state (data part) straight from registers
#pragma unroll
    for (int j = 0; j < PPT; ++j)
      if (cc[j] >= 0 && cc[j] < NDATA)
        *(float2*)(out + obase[j] + (long)(stp + 1) * NDATA) =
            make_float2(yv[j][0], yv[j][1]);
  }
}

extern "C" void kernel_launch(void* const* d_in, const int* in_sizes, int n_in,
                              void* d_out, int out_size, void* d_ws, size_t ws_size,
                              hipStream_t stream) {
  (void)in_sizes; (void)n_in; (void)out_size; (void)ws_size;
  const float* ts  = (const float*)d_in[0];
  const float* xs  = (const float*)d_in[1];
  const float* a_s = (const float*)d_in[2];
  const float* W0  = (const float*)d_in[3];
  const float* b0  = (const float*)d_in[4];
  const float* W1  = (const float*)d_in[5];
  const float* b1  = (const float*)d_in[6];
  const float* W2  = (const float*)d_in[7];
  const float* b2  = (const float*)d_in[8];
  const float* W3  = (const float*)d_in[9];
  const float* b3  = (const float*)d_in[10];
  _Float16* wh = (_Float16*)d_ws;

  prep_weights<<<(WH_TOTAL + 255) / 256, 256, 0, stream>>>(W0, W1, W2, W3, wh);
  node_kernel<<<NB / ROWS, THREADS, 0, stream>>>(ts, xs, a_s, b0, b1, b2, b3, wh,
                                                 (float*)d_out);
}
